// GGNN_89421219103172
// MI455X (gfx1250) — matrix operand order repaired and checked by reading, not compiled, not run on hardware
//
#include <hip/hip_runtime.h>
#include <math.h>

// GGNN: B=8, N=1024, H=128, E=4, STEPS=3
#define B_     8
#define N_     1024
#define H_     128
#define E_     4
#define STEPS_ 3

typedef __attribute__((ext_vector_type(16))) _Float16 v16h;
typedef __attribute__((ext_vector_type(8)))  _Float16 v8h;
typedef __attribute__((ext_vector_type(8)))  float    v8f;

// ---------------------------------------------------------------------------
// WMMA fragment helpers (v_wmma_f32_16x16x32_f16, wave32).
// A (16x32, MxK), X row-major, leading dim lda:
//   lane<16 : row m0+lane,    K = {k0..k0+7, k0+16..k0+23}
//   lane>=16: row m0+lane-16, K = {k0+8..k0+15, k0+24..k0+31}
__device__ __forceinline__ v16h ld_afrag(const _Float16* __restrict__ X, int lda,
                                         int m0, int k0, unsigned lane) {
  int row = m0 + (int)(lane & 15u);
  int kb  = k0 + (int)((lane >> 4) << 3);
  const _Float16* p = X + (size_t)row * lda + kb;
  v8h lo = *(const v8h*)(p);
  v8h hi = *(const v8h*)(p + 16);
  return __builtin_shufflevector(lo, hi, 0,1,2,3,4,5,6,7,8,9,10,11,12,13,14,15);
}

// B (32x16, KxN) with B[k,n] = W[n,k], W row-major (N x K), leading dim ldb:
//   lane<16 : N = n0+lane,    K = {k0..k0+7, k0+16..k0+23}
//   lane>=16: N = n0+lane-16, K = {k0+8..k0+15, k0+24..k0+31}
__device__ __forceinline__ v16h ld_bfrag(const _Float16* __restrict__ W, int ldb,
                                         int n0, int k0, unsigned lane) {
  int col = n0 + (int)(lane & 15u);
  int kb  = k0 + (int)((lane >> 4) << 3);
  const _Float16* p = W + (size_t)col * ldb + kb;
  v8h lo = *(const v8h*)(p);
  v8h hi = *(const v8h*)(p + 16);
  return __builtin_shufflevector(lo, hi, 0,1,2,3,4,5,6,7,8,9,10,11,12,13,14,15);
}

// C/D (16x16 f32): VGPR r holds M = r (+8 for upper half-wave), N = lane&15.
__device__ __forceinline__ void st_cfrag_f32(float* __restrict__ C, int ldc,
                                             int m0, int n0, unsigned lane,
                                             const v8f& acc) {
  int rbase = m0 + (int)((lane >> 4) << 3);
  int col   = n0 + (int)(lane & 15u);
#pragma unroll
  for (int r = 0; r < 8; ++r)
    C[(size_t)(rbase + r) * ldc + col] = acc[r];
}

__device__ __forceinline__ void st_cfrag_f16(_Float16* __restrict__ C, int ldc,
                                             int m0, int n0, unsigned lane,
                                             const v8f& acc) {
  int rbase = m0 + (int)((lane >> 4) << 3);
  int col   = n0 + (int)(lane & 15u);
#pragma unroll
  for (int r = 0; r < 8; ++r)
    C[(size_t)(rbase + r) * ldc + col] = (_Float16)acc[r];
}

#define WMMA_F16(acc, a, b) \
  (acc) = __builtin_amdgcn_wmma_f32_16x16x32_f16(false, (a), false, (b), (short)0, (acc), false, false)

// ---------------------------------------------------------------------------
// 32x64 wave tile: 2 A-frags x 4 B-frags -> 8 WMMAs per 32-deep K step.
// Double-buffered (f0/f1) so loads for the next K step overlap WMMAs.
struct Frag6 { v16h a0, a1, b0, b1, b2, b3; };

__device__ __forceinline__ void ld_tile(Frag6& f,
                                        const _Float16* __restrict__ A, int lda, int m0,
                                        const _Float16* __restrict__ W, int ldb, int n0,
                                        int k, unsigned lane) {
  f.a0 = ld_afrag(A, lda, m0,      k, lane);
  f.a1 = ld_afrag(A, lda, m0 + 16, k, lane);
  f.b0 = ld_bfrag(W, ldb, n0,      k, lane);
  f.b1 = ld_bfrag(W, ldb, n0 + 16, k, lane);
  f.b2 = ld_bfrag(W, ldb, n0 + 32, k, lane);
  f.b3 = ld_bfrag(W, ldb, n0 + 48, k, lane);
}

__device__ __forceinline__ void mma_tile(v8f acc[8], const Frag6& f) {
  WMMA_F16(acc[0], f.a0, f.b0);
  WMMA_F16(acc[1], f.a0, f.b1);
  WMMA_F16(acc[2], f.a0, f.b2);
  WMMA_F16(acc[3], f.a0, f.b3);
  WMMA_F16(acc[4], f.a1, f.b0);
  WMMA_F16(acc[5], f.a1, f.b1);
  WMMA_F16(acc[6], f.a1, f.b2);
  WMMA_F16(acc[7], f.a1, f.b3);
}

__device__ __forceinline__ void st_tile_f16(_Float16* __restrict__ C, int ldc,
                                            int m0, int n0, unsigned lane,
                                            const v8f acc[8]) {
#pragma unroll
  for (int mi = 0; mi < 2; ++mi)
#pragma unroll
    for (int t = 0; t < 4; ++t)
      st_cfrag_f16(C, ldc, m0 + mi * 16, n0 + t * 16, lane, acc[mi * 4 + t]);
}

__device__ __forceinline__ void st_tile_f32(float* __restrict__ C, int ldc,
                                            int m0, int n0, unsigned lane,
                                            const v8f acc[8]) {
#pragma unroll
  for (int mi = 0; mi < 2; ++mi)
#pragma unroll
    for (int t = 0; t < 4; ++t)
      st_cfrag_f32(C, ldc, m0 + mi * 16, n0 + t * 16, lane, acc[mi * 4 + t]);
}

// ---------------------------------------------------------------------------
// Kernel 1: per-edge-type linear, emitted TRANSPOSED:
//   tT[b,e][k][j] = sum_h We[e][k][h] * h16[b][j][h]
// grid: (N/64, 1, B*E), block 128 (4 waves, each 32x64; WG covers 128x64).
__global__ void ggnn_edge_linear_T(const _Float16* __restrict__ We16,
                                   const _Float16* __restrict__ h16,
                                   _Float16* __restrict__ tT) {
  unsigned lane = threadIdx.x & 31u;
  unsigned wave = threadIdx.x >> 5;
  int z = blockIdx.z;                 // b*E + e
  int e = z & (E_ - 1);
  int b = z >> 2;
  const _Float16* A = We16 + (size_t)e * H_ * H_;
  const _Float16* W = h16 + (size_t)b * N_ * H_;
  _Float16* C = tT + (size_t)z * H_ * N_;

  int m0 = (blockIdx.y * 4 + wave) * 32;   // output row (k of t), 0..127
  int n0 = blockIdx.x * 64;                // output col (node j)
  v8f acc[8] = {};
  Frag6 f0, f1;
  ld_tile(f0, A, H_, m0, W, H_, n0, 0, lane);
  for (int k = 0; k + 64 <= H_; k += 64) {
    ld_tile(f1, A, H_, m0, W, H_, n0, k + 32, lane);
    mma_tile(acc, f0);
    if (k + 64 < H_) ld_tile(f0, A, H_, m0, W, H_, n0, k + 64, lane);
    mma_tile(acc, f1);
  }
  st_tile_f16(C, N_, m0, n0, lane, acc);
}

// ---------------------------------------------------------------------------
// Kernel 2: message aggregation (dense SpMM over all edge types):
//   m16[b][i][k] = sum_e sum_j adj16[b,e][i][j] * tT[b,e][k][j]
// grid: (H/64, N/128, B), block 128. K total = E*N = 4096.
__global__ void ggnn_spmm(const _Float16* __restrict__ adj16,
                          const _Float16* __restrict__ tT,
                          _Float16* __restrict__ m16) {
  unsigned lane = threadIdx.x & 31u;
  unsigned wave = threadIdx.x >> 5;
  int b = blockIdx.z;
  int m0 = (blockIdx.y * 4 + wave) * 32;   // node i
  int n0 = blockIdx.x * 64;                // feature k
  v8f acc[8] = {};
  for (int e = 0; e < E_; ++e) {
    const _Float16* A = adj16 + ((size_t)(b * E_ + e)) * N_ * N_;
    const _Float16* W = tT + ((size_t)(b * E_ + e)) * H_ * N_;
    Frag6 f0, f1;
    ld_tile(f0, A, N_, m0, W, N_, n0, 0, lane);
    for (int k = 0; k + 64 <= N_; k += 64) {
      // pull the streaming adjacency ahead of the double buffer (speculative)
      if (k + 256 < N_)
        __builtin_prefetch((const void*)(A + (size_t)(m0 + (int)(lane & 15u)) * N_ + k + 256), 0, 0);
      ld_tile(f1, A, N_, m0, W, N_, n0, k + 32, lane);
      mma_tile(acc, f0);
      if (k + 64 < N_) ld_tile(f0, A, N_, m0, W, N_, n0, k + 64, lane);
      mma_tile(acc, f1);
    }
  }
  _Float16* C = m16 + (size_t)b * N_ * H_;
  st_tile_f16(C, H_, m0, n0, lane, acc);
}

// ---------------------------------------------------------------------------
// Kernel 3: generic C(f32) = X(f16, MxK) @ W(f16, NoutxK)^T
// grid: (Nout/64, M/128), block 128.
__global__ void ggnn_gemm_xwT(const _Float16* __restrict__ X, int lda,
                              const _Float16* __restrict__ W, int ldb,
                              float* __restrict__ C, int ldc, int K) {
  unsigned lane = threadIdx.x & 31u;
  unsigned wave = threadIdx.x >> 5;
  int m0 = (blockIdx.y * 4 + wave) * 32;
  int n0 = blockIdx.x * 64;
  v8f acc[8] = {};
  Frag6 f0, f1;
  ld_tile(f0, X, lda, m0, W, ldb, n0, 0, lane);
  for (int k = 0; k + 64 <= K; k += 64) {
    ld_tile(f1, X, lda, m0, W, ldb, n0, k + 32, lane);
    mma_tile(acc, f0);
    if (k + 64 < K) ld_tile(f0, X, lda, m0, W, ldb, n0, k + 64, lane);
    mma_tile(acc, f1);
  }
  st_tile_f32(C, ldc, m0, n0, lane, acc);
}

// ---------------------------------------------------------------------------
// Elementwise kernels
__global__ void ggnn_cvt_f32_f16(const float* __restrict__ s,
                                 _Float16* __restrict__ d, int n) {
  int i = blockIdx.x * blockDim.x + threadIdx.x;
  int stride = gridDim.x * blockDim.x;
  for (; i < n; i += stride) d[i] = (_Float16)s[i];
}

// GRU gates: gi/gh are (B*N, 3H) raw matmul outputs; biases added here.
__global__ void ggnn_gru_elem(const float* __restrict__ gi, const float* __restrict__ gh,
                              const float* __restrict__ b_ih, const float* __restrict__ b_hh,
                              float* __restrict__ h32, _Float16* __restrict__ h16,
                              int total) {
  int i = blockIdx.x * blockDim.x + threadIdx.x;
  if (i >= total) return;
  int row = i >> 7;           // / H_
  int k   = i & (H_ - 1);
  size_t base = (size_t)row * (3 * H_);
  float i_r = gi[base + k]           + b_ih[k];
  float i_z = gi[base + H_ + k]      + b_ih[H_ + k];
  float i_n = gi[base + 2 * H_ + k]  + b_ih[2 * H_ + k];
  float h_r = gh[base + k]           + b_hh[k];
  float h_z = gh[base + H_ + k]      + b_hh[H_ + k];
  float h_n = gh[base + 2 * H_ + k]  + b_hh[2 * H_ + k];
  float r = 1.0f / (1.0f + __expf(-(i_r + h_r)));
  float z = 1.0f / (1.0f + __expf(-(i_z + h_z)));
  float n = tanhf(i_n + r * h_n);
  float h = h32[i];
  float hn = (1.0f - z) * n + z * h;
  h32[i] = hn;
  h16[i] = (_Float16)hn;
}

__global__ void ggnn_build_cat(const float* __restrict__ x0, const float* __restrict__ h32,
                               _Float16* __restrict__ cat16, int total) {
  int i = blockIdx.x * blockDim.x + threadIdx.x;
  if (i >= total) return;
  int row = i >> 7;
  int k   = i & (H_ - 1);
  cat16[(size_t)row * (2 * H_) + k]      = (_Float16)x0[i];
  cat16[(size_t)row * (2 * H_) + H_ + k] = (_Float16)h32[i];
}

__global__ void ggnn_relu_bias(const float* __restrict__ x, const float* __restrict__ bias,
                               _Float16* __restrict__ y, int total) {
  int i = blockIdx.x * blockDim.x + threadIdx.x;
  if (i >= total) return;
  float v = x[i] + bias[i & (H_ - 1)];
  y[i] = (_Float16)fmaxf(v, 0.0f);
}

__global__ void ggnn_add_bias(const float* __restrict__ x, const float* __restrict__ bias,
                              float* __restrict__ y, int total) {
  int i = blockIdx.x * blockDim.x + threadIdx.x;
  if (i >= total) return;
  y[i] = x[i] + bias[i & (H_ - 1)];
}

// ---------------------------------------------------------------------------
extern "C" void kernel_launch(void* const* d_in, const int* in_sizes, int n_in,
                              void* d_out, int out_size, void* d_ws, size_t ws_size,
                              hipStream_t stream) {
  (void)in_sizes; (void)n_in; (void)out_size; (void)ws_size;

  const float* node_features = (const float*)d_in[0];  // B,N,H
  const float* edge_matrix   = (const float*)d_in[1];  // B,E,N,N
  const float* We            = (const float*)d_in[2];  // E,H,H
  const float* W_ih          = (const float*)d_in[3];  // 3H,H
  const float* W_hh          = (const float*)d_in[4];  // 3H,H
  const float* b_ih          = (const float*)d_in[5];  // 3H
  const float* b_hh          = (const float*)d_in[6];  // 3H
  const float* W1            = (const float*)d_in[7];  // H,2H
  const float* b1            = (const float*)d_in[8];  // H
  const float* W2            = (const float*)d_in[9];  // H,H
  const float* b2            = (const float*)d_in[10]; // H
  float* out = (float*)d_out;

  // ---- workspace layout ----
  size_t off = 0;
  auto alloc = [&](size_t bytes) -> void* {
    void* p = (char*)d_ws + off;
    off += (bytes + 255) & ~(size_t)255;
    return p;
  };
  const int nBNH  = B_ * N_ * H_;          // 1,048,576
  const int nADJ  = B_ * E_ * N_ * N_;     // 33,554,432
  const int nT    = B_ * E_ * H_ * N_;     // 4,194,304
  const int nG    = B_ * N_ * 3 * H_;      // 3,145,728

  _Float16* adj16 = (_Float16*)alloc((size_t)nADJ * 2);
  _Float16* tT16  = (_Float16*)alloc((size_t)nT * 2);
  _Float16* h16   = (_Float16*)alloc((size_t)nBNH * 2);
  float*    h32   = (float*)   alloc((size_t)nBNH * 4);
  _Float16* m16   = (_Float16*)alloc((size_t)nBNH * 2);
  float*    gi32  = (float*)   alloc((size_t)nG * 4);
  float*    gh32  = (float*)   alloc((size_t)nG * 4);
  _Float16* We16  = (_Float16*)alloc((size_t)E_ * H_ * H_ * 2);
  _Float16* Wih16 = (_Float16*)alloc((size_t)3 * H_ * H_ * 2);
  _Float16* Whh16 = (_Float16*)alloc((size_t)3 * H_ * H_ * 2);
  _Float16* W116  = (_Float16*)alloc((size_t)H_ * 2 * H_ * 2);
  _Float16* W216  = (_Float16*)alloc((size_t)H_ * H_ * 2);
  // MLP scratch overlaid on the (then-dead) gi/gh region to trim footprint:
  _Float16* cat16 = (_Float16*)gi32;                       // 8192*256 f16 = 4 MB
  float*    hid32 = (float*)((char*)gi32 + (((size_t)nBNH * 2 * 2 + 255) & ~(size_t)255));
  float*    y32   = gh32;                                  // 8192*128 f32 = 4 MB
  _Float16* hid16 = m16;                                   // reuse m16 (dead after steps)

  const dim3 blk(128);
  const int ET = 256;                                      // elementwise block

  // ---- conversions (re-done every call; deterministic) ----
  ggnn_cvt_f32_f16<<<dim3((E_*H_*H_ + ET-1)/ET), ET, 0, stream>>>(We, We16, E_*H_*H_);
  ggnn_cvt_f32_f16<<<dim3((3*H_*H_ + ET-1)/ET), ET, 0, stream>>>(W_ih, Wih16, 3*H_*H_);
  ggnn_cvt_f32_f16<<<dim3((3*H_*H_ + ET-1)/ET), ET, 0, stream>>>(W_hh, Whh16, 3*H_*H_);
  ggnn_cvt_f32_f16<<<dim3((2*H_*H_ + ET-1)/ET), ET, 0, stream>>>(W1, W116, 2*H_*H_);
  ggnn_cvt_f32_f16<<<dim3((H_*H_ + ET-1)/ET), ET, 0, stream>>>(W2, W216, H_*H_);
  ggnn_cvt_f32_f16<<<dim3((nADJ + ET-1)/ET), ET, 0, stream>>>(edge_matrix, adj16, nADJ);
  ggnn_cvt_f32_f16<<<dim3((nBNH + ET-1)/ET), ET, 0, stream>>>(node_features, h16, nBNH);
  hipMemcpyAsync(h32, node_features, (size_t)nBNH * 4, hipMemcpyDeviceToDevice, stream);

  // ---- propagation steps ----
  for (int s = 0; s < STEPS_; ++s) {
    // tT[b,e] = We[e] @ h[b]^T        (M=128, N=1024, K=128) x 32 batches
    ggnn_edge_linear_T<<<dim3(N_/64, 1, B_*E_), blk, 0, stream>>>(We16, h16, tT16);
    // m[b] = sum_e adj[b,e] @ t[b,e]  (M=1024, N=128, K=4096) x 8 batches
    ggnn_spmm<<<dim3(H_/64, N_/128, B_), blk, 0, stream>>>(adj16, tT16, m16);
    // GRU gate matmuls: (8192 x 384) = (8192 x 128) @ (384 x 128)^T
    ggnn_gemm_xwT<<<dim3(3*H_/64, (B_*N_)/128), blk, 0, stream>>>(m16, H_, Wih16, H_, gi32, 3*H_, H_);
    ggnn_gemm_xwT<<<dim3(3*H_/64, (B_*N_)/128), blk, 0, stream>>>(h16, H_, Whh16, H_, gh32, 3*H_, H_);
    // gate nonlinearities + state update
    ggnn_gru_elem<<<dim3(nBNH/ET), ET, 0, stream>>>(gi32, gh32, b_ih, b_hh, h32, h16, nBNH);
  }

  // ---- output head: y = relu([x0, h] @ W1^T + b1) @ W2^T + b2 ----
  ggnn_build_cat<<<dim3(nBNH/ET), ET, 0, stream>>>(node_features, h32, cat16, nBNH);
  ggnn_gemm_xwT<<<dim3(H_/64, (B_*N_)/128), blk, 0, stream>>>(cat16, 2*H_, W116, 2*H_, hid32, H_, 2*H_);
  ggnn_relu_bias<<<dim3(nBNH/ET), ET, 0, stream>>>(hid32, b1, hid16, nBNH);
  ggnn_gemm_xwT<<<dim3(H_/64, (B_*N_)/128), blk, 0, stream>>>(hid16, H_, W216, H_, y32, H_, H_);
  ggnn_add_bias<<<dim3(nBNH/ET), ET, 0, stream>>>(y32, b2, out, nBNH);
}
